// Model_VFPT_TCPA_46935402610874
// MI455X (gfx1250) — matrix operand order repaired and checked by reading, not compiled, not run on hardware
//
#include <hip/hip_runtime.h>
#include <math.h>

// ---------------------------------------------------------------------------
// ViT-B/16 VPT forward for MI455X (gfx1250):
//  - all GEMMs via v_wmma_f32_16x16x32_bf16 (fp32 accumulate)
//  - LDS staging via global_load_async_to_lds_b128 (ASYNCcnt), double-buffered
// Requires ws_size >= ~540MB.
// ---------------------------------------------------------------------------

typedef __bf16 bf16;
typedef __attribute__((ext_vector_type(16))) __bf16 v16bf;
typedef __attribute__((ext_vector_type(8)))  float  v8f;

#define WMMA_BF16(a, b, c) \
  __builtin_amdgcn_wmma_f32_16x16x32_bf16(false, (a), false, (b), (short)0, (c), false, false)

#define BATCH   64
#define LAYERS  12
#define HEADS   12
#define DHEAD   64
#define DMODEL  768
#define FFDIM   3072
#define PLEN    20
#define NTOK    197     // kept tokens
#define NFULL   217     // tokens with prompts
#define QPAD    208     // 13*16
#define KPAD    224     // 14*16
#define NCLS    100

// ---------------------------------------------------------------------------
// Async global -> LDS copy of 16 bytes (per active lane), tracked by ASYNCcnt.
// LDS byte address = low 32 bits of the flat pointer (ISA 10.2: LDS aperture).
__device__ __forceinline__ void async_copy16(const void* gsrc, void* ldst) {
  unsigned lds_addr = (unsigned)(size_t)ldst;
  asm volatile("global_load_async_to_lds_b128 %0, %1, off"
               :: "v"(lds_addr), "v"(gsrc) : "memory");
}
__device__ __forceinline__ void wait_async0() {
  asm volatile("s_wait_asynccnt 0x0" ::: "memory");
}
__device__ __forceinline__ void lds_zero16(void* ldst) {
  *(uint4*)ldst = make_uint4(0u, 0u, 0u, 0u);
}

// ---------------------------------------------------------------------------
// fp32 -> bf16 conversion
__global__ __launch_bounds__(256) void cvt_bf16(const float* __restrict__ in,
                                                bf16* __restrict__ out, int n) {
  int i = blockIdx.x * 256 + threadIdx.x;
  if (i < n) out[i] = (bf16)in[i];
}

// ---------------------------------------------------------------------------
// 2D DFT of prompts: fft over D (768) then over P (20), keep real part.
__global__ __launch_bounds__(256) void fft_stage1(const float* __restrict__ prompts,
                                                  float* __restrict__ f1r,
                                                  float* __restrict__ f1i) {
  int idx = blockIdx.x * 256 + threadIdx.x;
  if (idx >= LAYERS * PLEN * DMODEL) return;
  int d = idx % DMODEL;
  int rest = idx / DMODEL;
  const float* src = prompts + (size_t)rest * DMODEL;  // row (l,p)
  float re = 0.f, im = 0.f;
  for (int dp = 0; dp < DMODEL; ++dp) {
    int t = (d * dp) % DMODEL;
    float th = 6.28318530717958647692f * (float)t / (float)DMODEL;
    float s, c;
    __sincosf(th, &s, &c);
    float x = src[dp];
    re += x * c;
    im -= x * s;
  }
  f1r[idx] = re;
  f1i[idx] = im;
}

__global__ __launch_bounds__(256) void fft_stage2(const float* __restrict__ f1r,
                                                  const float* __restrict__ f1i,
                                                  float* __restrict__ pr) {
  int idx = blockIdx.x * 256 + threadIdx.x;
  if (idx >= LAYERS * PLEN * DMODEL) return;
  int d = idx % DMODEL;
  int rest = idx / DMODEL;
  int p = rest % PLEN;
  int l = rest / PLEN;
  float acc = 0.f;
  for (int pp = 0; pp < PLEN; ++pp) {
    int t = (p * pp) % PLEN;
    float th = 6.28318530717958647692f * (float)t / (float)PLEN;
    float s, c;
    __sincosf(th, &s, &c);
    size_t si = ((size_t)l * PLEN + pp) * DMODEL + d;
    acc += f1r[si] * c + f1i[si] * s;  // Re((Fr+iFi)*(c - i s))
  }
  pr[idx] = acc;
}

// ---------------------------------------------------------------------------
// im2col: x[B,3,224,224] -> pimg bf16 [B*196, 768], col = (c,py,px)
__global__ __launch_bounds__(256) void im2col_k(const float* __restrict__ x,
                                                bf16* __restrict__ pimg) {
  int idx = blockIdx.x * 256 + threadIdx.x;  // B*196*768, exact multiple of 256
  int d = idx % DMODEL;
  int rest = idx / DMODEL;
  int pidx = rest % 196;
  int b = rest / 196;
  int gy = pidx / 14, gx = pidx % 14;
  int c = d / 256, r2 = d % 256;
  int py = r2 / 16, px = r2 % 16;
  float v = x[(((size_t)b * 3 + c) * 224 + gy * 16 + py) * 224 + gx * 16 + px];
  pimg[(size_t)idx] = (bf16)v;
}

// assemble h[B,197,768]: row0 = cls + pos0 ; rows 1.. = pemb + pos
__global__ __launch_bounds__(256) void assemble_h(const float* __restrict__ pemb,
                                                  const float* __restrict__ cls_tok,
                                                  const float* __restrict__ pos,
                                                  float* __restrict__ h) {
  int idx = blockIdx.x * 256 + threadIdx.x;  // B*197*768, exact multiple of 256
  int d = idx % DMODEL;
  int rest = idx / DMODEL;
  int r = rest % NTOK;
  int b = rest / NTOK;
  float v;
  if (r == 0)
    v = cls_tok[d] + pos[d];
  else
    v = pemb[((size_t)b * 196 + (r - 1)) * DMODEL + d] + pos[(size_t)r * DMODEL + d];
  h[(size_t)idx] = v;
}

// ---------------------------------------------------------------------------
// LayerNorm -> bf16. mode 0: virtual xi rows (row0=h0, 1..20=pr, 21..=h[r-20]);
// mode 1: plain rows of h.
__global__ __launch_bounds__(256) void ln_kernel(const float* __restrict__ h,
                                                 const float* __restrict__ pr,
                                                 const float* __restrict__ g,
                                                 const float* __restrict__ bta,
                                                 bf16* __restrict__ out, int mode) {
  __shared__ float sred[256];
  int row = blockIdx.x;
  int tid = threadIdx.x;
  const float* src;
  if (mode == 0) {
    int b = row / NFULL, r = row % NFULL;
    if (r == 0)         src = h + (size_t)b * NTOK * DMODEL;
    else if (r <= PLEN) src = pr + (size_t)(r - 1) * DMODEL;
    else                src = h + ((size_t)b * NTOK + (r - PLEN)) * DMODEL;
  } else {
    src = h + (size_t)row * DMODEL;
  }
  float xv[3];
  float s1 = 0.f, s2 = 0.f;
#pragma unroll
  for (int k = 0; k < 3; ++k) {
    xv[k] = src[tid + k * 256];
    s1 += xv[k];
    s2 += xv[k] * xv[k];
  }
  sred[tid] = s1; __syncthreads();
  for (int off = 128; off; off >>= 1) { if (tid < off) sred[tid] += sred[tid + off]; __syncthreads(); }
  float mean = sred[0] * (1.0f / DMODEL); __syncthreads();
  sred[tid] = s2; __syncthreads();
  for (int off = 128; off; off >>= 1) { if (tid < off) sred[tid] += sred[tid + off]; __syncthreads(); }
  float var = sred[0] * (1.0f / DMODEL) - mean * mean;
  float inv = rsqrtf(var + 1e-6f);
  bf16* dst = out + (size_t)row * DMODEL;
#pragma unroll
  for (int k = 0; k < 3; ++k) {
    int i = tid + k * 256;
    dst[i] = (bf16)((xv[k] - mean) * inv * g[i] + bta[i]);
  }
}

// ---------------------------------------------------------------------------
// Tiled WMMA GEMM, BM=128 BN=128 BK=32, double-buffered async LDS staging.
// ep 0: fp32 out = acc+bias
// ep 1: fp32 out = res + acc + bias
// ep 2: bf16 out = gelu(acc+bias)
// ep 3: bf16 out = acc+bias
__global__ __launch_bounds__(256) void gemm_bf16(const bf16* __restrict__ A, int lda,
                                                 const bf16* __restrict__ Bm, int ldb,
                                                 const float* __restrict__ bias,
                                                 const float* __restrict__ res,
                                                 float* __restrict__ outF,
                                                 bf16* __restrict__ outB,
                                                 int M, int N, int K, int ep) {
  __shared__ bf16 As[2][128][32];
  __shared__ bf16 Bs[2][32][128];
  int tid = threadIdx.x;
  int lane = tid & 31, wid = tid >> 5;
  int wm = wid >> 1, wn = wid & 1;        // 4 x 2 wave grid; wave owns 32x64
  int bm = blockIdx.x, bn = blockIdx.y;

  v8f zero = {0.f, 0.f, 0.f, 0.f, 0.f, 0.f, 0.f, 0.f};
  v8f acc[2][4];
#pragma unroll
  for (int i = 0; i < 2; ++i)
#pragma unroll
    for (int j = 0; j < 4; ++j) acc[i][j] = zero;

  auto stage = [&](int buf, int kt) {
    // A tile [128 x 32]: 512 chunks of 8 bf16
#pragma unroll
    for (int c = tid; c < 512; c += 256) {
      int r = c >> 2, kc = (c & 3) << 3;
      int gm = bm * 128 + r;
      if (gm < M)
        async_copy16(A + (size_t)gm * lda + kt * 32 + kc, &As[buf][r][kc]);
      else
        lds_zero16(&As[buf][r][kc]);
    }
    // B tile [32 x 128]: 512 chunks of 8 bf16 (N always multiple of 128)
#pragma unroll
    for (int c = tid; c < 512; c += 256) {
      int r = c >> 4, nc = (c & 15) << 3;
      async_copy16(Bm + (size_t)(kt * 32 + r) * ldb + bn * 128 + nc, &Bs[buf][r][nc]);
    }
  };

  int nkt = K >> 5;
  int kb = (lane & 16) ? 8 : 0;
  int mrow0 = wm * 32 + (lane & 15);

  stage(0, 0);
  int cur = 0;
  for (int kt = 0; kt < nkt; ++kt) {
    wait_async0();
    __syncthreads();
    if (kt + 1 < nkt) stage(cur ^ 1, kt + 1);

    v16bf a0, a1;
#pragma unroll
    for (int j = 0; j < 8; ++j) {
      a0[j]     = As[cur][mrow0][kb + j];
      a0[j + 8] = As[cur][mrow0][kb + 16 + j];
      a1[j]     = As[cur][mrow0 + 16][kb + j];
      a1[j + 8] = As[cur][mrow0 + 16][kb + 16 + j];
    }
    v16bf bfr[4];
#pragma unroll
    for (int t = 0; t < 4; ++t)
#pragma unroll
      for (int j = 0; j < 16; ++j)
        bfr[t][j] = Bs[cur][kb + (j & 7) + ((j >> 3) << 4)][wn * 64 + t * 16 + (lane & 15)];

#pragma unroll
    for (int t = 0; t < 4; ++t) {
      acc[0][t] = WMMA_BF16(a0, bfr[t], acc[0][t]);
      acc[1][t] = WMMA_BF16(a1, bfr[t], acc[1][t]);
    }
    cur ^= 1;
  }

#pragma unroll
  for (int ti = 0; ti < 2; ++ti)
#pragma unroll
    for (int tj = 0; tj < 4; ++tj)
#pragma unroll
      for (int r = 0; r < 8; ++r) {
        int m = bm * 128 + wm * 32 + ti * 16 + r + ((lane & 16) ? 8 : 0);
        int n = bn * 128 + wn * 64 + tj * 16 + (lane & 15);
        if (m >= M) continue;
        float v = acc[ti][tj][r];
        if (bias) v += bias[n];
        size_t oi = (size_t)m * N + n;
        if (ep == 1) {
          outF[oi] = res[oi] + v;
        } else if (ep == 2) {
          outB[oi] = (bf16)(0.5f * v * (1.0f + erff(v * 0.70710678118654752f)));
        } else if (ep == 3) {
          outB[oi] = (bf16)v;
        } else {
          outF[oi] = v;
        }
      }
}

// ---------------------------------------------------------------------------
// Attention scores per (b,h): S[QPAD,KPAD] = 0.125 * Qkept @ K^T  (qkv is bf16)
__global__ __launch_bounds__(256) void attn_scores(const bf16* __restrict__ qkv,
                                                   float* __restrict__ scores) {
  __shared__ bf16 Qs[QPAD][DHEAD];
  __shared__ bf16 Ks[KPAD][DHEAD];
  int b = blockIdx.x / HEADS, hh = blockIdx.x % HEADS;
  int tid = threadIdx.x, lane = tid & 31, wid = tid >> 5;

  // stage Q (kept-row gather) and K via async engine; pad rows zeroed
  for (int c = tid; c < (QPAD * DHEAD) / 8; c += 256) {
    int r = c >> 3, dc = (c & 7) << 3;
    if (r < NTOK) {
      int sr = (r == 0) ? 0 : r + PLEN;  // kept query rows of xi
      async_copy16(qkv + ((size_t)(b * NFULL + sr)) * 2304 + hh * 64 + dc, &Qs[r][dc]);
    } else {
      lds_zero16(&Qs[r][dc]);
    }
  }
  for (int c = tid; c < (KPAD * DHEAD) / 8; c += 256) {
    int r = c >> 3, dc = (c & 7) << 3;
    if (r < NFULL)
      async_copy16(qkv + ((size_t)(b * NFULL + r)) * 2304 + 768 + hh * 64 + dc, &Ks[r][dc]);
    else
      lds_zero16(&Ks[r][dc]);
  }
  wait_async0();
  __syncthreads();

  size_t sbase = (size_t)blockIdx.x * QPAD * KPAD;
  int kb = (lane & 16) ? 8 : 0;
  for (int t = wid; t < 13 * 14; t += 8) {
    int tm = t / 14, tn = t % 14;
    v8f acc = {0.f, 0.f, 0.f, 0.f, 0.f, 0.f, 0.f, 0.f};
    int mr = tm * 16 + (lane & 15);
#pragma unroll
    for (int kk = 0; kk < DHEAD; kk += 32) {
      v16bf a, bfr;
#pragma unroll
      for (int j = 0; j < 8; ++j) {
        a[j]     = Qs[mr][kk + kb + j];
        a[j + 8] = Qs[mr][kk + kb + 16 + j];
      }
#pragma unroll
      for (int j = 0; j < 16; ++j)
        bfr[j] = Ks[tn * 16 + (lane & 15)][kk + kb + (j & 7) + ((j >> 3) << 4)];  // B[k][n] = K[n][k]
      acc = WMMA_BF16(a, bfr, acc);
    }
#pragma unroll
    for (int r = 0; r < 8; ++r) {
      int m = tm * 16 + r + ((lane & 16) ? 8 : 0);
      int n = tn * 16 + (lane & 15);
      scores[sbase + (size_t)m * KPAD + n] = acc[r] * 0.125f;
    }
  }
}

// softmax over 217 keys, then * mask[k]; write zero-padded bf16 probs
__global__ __launch_bounds__(256) void softmax_mask(const float* __restrict__ scores,
                                                    const float* __restrict__ mask,
                                                    bf16* __restrict__ probs) {
  int row = blockIdx.x;  // B*H*QPAD
  int r = row % QPAD;
  size_t base = (size_t)(row / QPAD) * QPAD * KPAD + (size_t)r * KPAD;
  int tid = threadIdx.x;
  if (r >= NTOK) {
    if (tid < KPAD) probs[base + tid] = (bf16)0.f;
    return;
  }
  __shared__ float sred[256];
  float v = (tid < NFULL) ? scores[base + tid] : -1e30f;
  sred[tid] = v; __syncthreads();
  for (int off = 128; off; off >>= 1) { if (tid < off) sred[tid] = fmaxf(sred[tid], sred[tid + off]); __syncthreads(); }
  float mx = sred[0]; __syncthreads();
  float e = (tid < NFULL) ? __expf(v - mx) : 0.f;
  sred[tid] = e; __syncthreads();
  for (int off = 128; off; off >>= 1) { if (tid < off) sred[tid] += sred[tid + off]; __syncthreads(); }
  float inv = 1.0f / sred[0];
  if (tid < KPAD) {
    float p = (tid < NFULL) ? e * inv * mask[tid] : 0.f;
    probs[base + tid] = (bf16)p;
  }
}

// O[197,64] = P[QPAD,KPAD] @ V[KPAD,64] per (b,h); ADVP affine on V prompt rows
__global__ __launch_bounds__(256) void attn_pv(const bf16* __restrict__ qkv,
                                               const bf16* __restrict__ probs,
                                               const float* __restrict__ advw,
                                               const float* __restrict__ advb,
                                               bf16* __restrict__ o) {
  __shared__ bf16 Vs[KPAD][DHEAD];
  int b = blockIdx.x / HEADS, hh = blockIdx.x % HEADS;
  int tid = threadIdx.x, lane = tid & 31, wid = tid >> 5;

  for (int c = tid; c < (KPAD * DHEAD) / 8; c += 256) {
    int r = c >> 3, dc = (c & 7) << 3;
    if (r < NFULL)
      async_copy16(qkv + ((size_t)(b * NFULL + r)) * 2304 + 1536 + hh * 64 + dc, &Vs[r][dc]);
    else
      lds_zero16(&Vs[r][dc]);
  }
  wait_async0();
  __syncthreads();
  // ADVP affine fix-up on prompt rows 1..20 (in LDS)
  for (int i = tid; i < PLEN * DHEAD; i += 256) {
    int r = 1 + (i >> 6), d = i & 63;
    int ai = (r - 1) * DMODEL + hh * 64 + d;
    Vs[r][d] = (bf16)((float)Vs[r][d] * advw[ai] + advb[ai]);
  }
  __syncthreads();

  size_t pbase = (size_t)blockIdx.x * QPAD * KPAD;
  int kb = (lane & 16) ? 8 : 0;
  for (int t = wid; t < 13 * 4; t += 8) {
    int tm = t >> 2, tn = t & 3;
    v8f acc = {0.f, 0.f, 0.f, 0.f, 0.f, 0.f, 0.f, 0.f};
    int mr = tm * 16 + (lane & 15);
    const bf16* prow = probs + pbase + (size_t)mr * KPAD;
#pragma unroll
    for (int kk = 0; kk < KPAD; kk += 32) {
      v16bf a, bfr;
      const bf16* p = prow + kk + kb;
#pragma unroll
      for (int j = 0; j < 8; ++j) { a[j] = p[j]; a[j + 8] = p[16 + j]; }
#pragma unroll
      for (int j = 0; j < 16; ++j)
        bfr[j] = Vs[kk + kb + (j & 7) + ((j >> 3) << 4)][tn * 16 + (lane & 15)];
      acc = WMMA_BF16(a, bfr, acc);
    }
#pragma unroll
    for (int r = 0; r < 8; ++r) {
      int m = tm * 16 + r + ((lane & 16) ? 8 : 0);
      int n = tn * 16 + (lane & 15);
      if (m < NTOK)
        o[((size_t)(b * NTOK + m)) * DMODEL + hh * 64 + n] = (bf16)acc[r];
    }
  }
}

// ---------------------------------------------------------------------------
// Final LN of cls row + head matvec (fp32; tiny)
__global__ __launch_bounds__(256) void head_kernel(const float* __restrict__ h,
                                                   const float* __restrict__ g,
                                                   const float* __restrict__ bta,
                                                   const float* __restrict__ hw,
                                                   const float* __restrict__ hb,
                                                   float* __restrict__ out) {
  __shared__ float sv[DMODEL];
  __shared__ float sred[256];
  int b = blockIdx.x, tid = threadIdx.x;
  const float* src = h + (size_t)b * NTOK * DMODEL;
  float xv[3], s1 = 0.f, s2 = 0.f;
#pragma unroll
  for (int k = 0; k < 3; ++k) {
    xv[k] = src[tid + k * 256];
    s1 += xv[k]; s2 += xv[k] * xv[k];
  }
  sred[tid] = s1; __syncthreads();
  for (int off = 128; off; off >>= 1) { if (tid < off) sred[tid] += sred[tid + off]; __syncthreads(); }
  float mean = sred[0] * (1.0f / DMODEL); __syncthreads();
  sred[tid] = s2; __syncthreads();
  for (int off = 128; off; off >>= 1) { if (tid < off) sred[tid] += sred[tid + off]; __syncthreads(); }
  float var = sred[0] * (1.0f / DMODEL) - mean * mean;
  float inv = rsqrtf(var + 1e-6f);
#pragma unroll
  for (int k = 0; k < 3; ++k) {
    int i = tid + k * 256;
    sv[i] = (xv[k] - mean) * inv * g[i] + bta[i];
  }
  __syncthreads();
  for (int c = tid; c < NCLS; c += 256) {
    float acc = hb[c];
    for (int d = 0; d < DMODEL; ++d) acc += sv[d] * hw[(size_t)d * NCLS + c];
    out[(size_t)b * NCLS + c] = acc;
  }
}

// ---------------------------------------------------------------------------
extern "C" void kernel_launch(void* const* d_in, const int* in_sizes, int n_in,
                              void* d_out, int out_size, void* d_ws, size_t ws_size,
                              hipStream_t stream) {
  (void)in_sizes; (void)n_in; (void)out_size; (void)ws_size;
  const float* x       = (const float*)d_in[0];
  const float* patch_w = (const float*)d_in[1];
  const float* patch_b = (const float*)d_in[2];
  const float* cls_tok = (const float*)d_in[3];
  const float* pos_emb = (const float*)d_in[4];
  const float* prompts = (const float*)d_in[5];
  const float* masks   = (const float*)d_in[6];
  const float* advp_w  = (const float*)d_in[7];
  const float* advp_b  = (const float*)d_in[8];
  const float* ln1_g   = (const float*)d_in[9];
  const float* ln1_b   = (const float*)d_in[10];
  const float* qkv_w   = (const float*)d_in[11];
  const float* qkv_b   = (const float*)d_in[12];
  const float* proj_w  = (const float*)d_in[13];
  const float* proj_b  = (const float*)d_in[14];
  const float* ln2_g   = (const float*)d_in[15];
  const float* ln2_b   = (const float*)d_in[16];
  const float* fc1_w   = (const float*)d_in[17];
  const float* fc1_b   = (const float*)d_in[18];
  const float* fc2_w   = (const float*)d_in[19];
  const float* fc2_b   = (const float*)d_in[20];
  const float* norm_g  = (const float*)d_in[21];
  const float* norm_b  = (const float*)d_in[22];
  const float* head_w  = (const float*)d_in[23];
  const float* head_b  = (const float*)d_in[24];
  float* out = (float*)d_out;

  char* ws = (char*)d_ws;
  size_t off = 0;
  auto alloc = [&](size_t bytes) -> void* {
    void* p = ws + off;
    off = (off + bytes + 255) & ~(size_t)255;
    return p;
  };

  const int MQ = BATCH * NFULL;   // 13888
  const int MT = BATCH * NTOK;    // 12608
  bf16*  wpatch = (bf16*)alloc((size_t)DMODEL * DMODEL * 2);
  bf16*  wqkv   = (bf16*)alloc((size_t)LAYERS * DMODEL * 3 * DMODEL * 2);
  bf16*  wproj  = (bf16*)alloc((size_t)LAYERS * DMODEL * DMODEL * 2);
  bf16*  wfc1   = (bf16*)alloc((size_t)LAYERS * DMODEL * FFDIM * 2);
  bf16*  wfc2   = (bf16*)alloc((size_t)LAYERS * FFDIM * DMODEL * 2);
  float* hbuf   = (float*)alloc((size_t)MT * DMODEL * 4);
  float* prb    = (float*)alloc((size_t)LAYERS * PLEN * DMODEL * 4);
  float* f1r    = (float*)alloc((size_t)LAYERS * PLEN * DMODEL * 4);
  float* f1i    = (float*)alloc((size_t)LAYERS * PLEN * DMODEL * 4);
  bf16*  ain    = (bf16*)alloc((size_t)MQ * DMODEL * 2);
  bf16*  qkvB   = (bf16*)alloc((size_t)MQ * 3 * DMODEL * 2);            // 64 MB
  float* scores = (float*)alloc((size_t)BATCH * HEADS * QPAD * KPAD * 4);
  bf16*  probs  = (bf16*)alloc((size_t)BATCH * HEADS * QPAD * KPAD * 2);
  bf16*  obuf   = (bf16*)alloc((size_t)MT * DMODEL * 2);
  // Aliases into dead regions (patch embed happens before qkv is first used):
  bf16*  pimg = (bf16*)qkvB;                                            // [12544,768] bf16
  float* pemb = (float*)((char*)qkvB + (((size_t)12544 * DMODEL * 2 + 255) & ~(size_t)255));
  bf16*  mlp  = (bf16*)scores;  // [12608,3072] bf16; scores dead before fc1

  // ---- one-time setup --------------------------------------------------
  int n;
  n = DMODEL * DMODEL;
  cvt_bf16<<<(n + 255) / 256, 256, 0, stream>>>(patch_w, wpatch, n);
  n = LAYERS * DMODEL * 3 * DMODEL;
  cvt_bf16<<<(n + 255) / 256, 256, 0, stream>>>(qkv_w, wqkv, n);
  n = LAYERS * DMODEL * DMODEL;
  cvt_bf16<<<(n + 255) / 256, 256, 0, stream>>>(proj_w, wproj, n);
  n = LAYERS * DMODEL * FFDIM;
  cvt_bf16<<<(n + 255) / 256, 256, 0, stream>>>(fc1_w, wfc1, n);
  n = LAYERS * FFDIM * DMODEL;
  cvt_bf16<<<(n + 255) / 256, 256, 0, stream>>>(fc2_w, wfc2, n);

  n = LAYERS * PLEN * DMODEL;
  fft_stage1<<<(n + 255) / 256, 256, 0, stream>>>(prompts, f1r, f1i);
  fft_stage2<<<(n + 255) / 256, 256, 0, stream>>>(f1r, f1i, prb);

  im2col_k<<<(BATCH * 196 * DMODEL) / 256, 256, 0, stream>>>(x, pimg);
  gemm_bf16<<<dim3(98, 6), 256, 0, stream>>>(pimg, DMODEL, wpatch, DMODEL, patch_b,
                                             nullptr, pemb, nullptr,
                                             BATCH * 196, DMODEL, DMODEL, 0);
  assemble_h<<<(BATCH * NTOK * DMODEL) / 256, 256, 0, stream>>>(pemb, cls_tok, pos_emb, hbuf);

  // ---- transformer layers ----------------------------------------------
  for (int l = 0; l < LAYERS; ++l) {
    ln_kernel<<<MQ, 256, 0, stream>>>(hbuf, prb + (size_t)l * PLEN * DMODEL,
                                      ln1_g + l * DMODEL, ln1_b + l * DMODEL, ain, 0);
    gemm_bf16<<<dim3(109, 18), 256, 0, stream>>>(ain, DMODEL,
        wqkv + (size_t)l * DMODEL * 3 * DMODEL, 3 * DMODEL,
        qkv_b + (size_t)l * 3 * DMODEL, nullptr, nullptr, qkvB,
        MQ, 3 * DMODEL, DMODEL, 3);
    attn_scores<<<BATCH * HEADS, 256, 0, stream>>>(qkvB, scores);
    softmax_mask<<<BATCH * HEADS * QPAD, 256, 0, stream>>>(scores, masks + (size_t)l * NFULL, probs);
    attn_pv<<<BATCH * HEADS, 256, 0, stream>>>(qkvB, probs, advp_w, advp_b, obuf);
    gemm_bf16<<<dim3(99, 6), 256, 0, stream>>>(obuf, DMODEL,
        wproj + (size_t)l * DMODEL * DMODEL, DMODEL,
        proj_b + (size_t)l * DMODEL, hbuf, hbuf, nullptr,
        MT, DMODEL, DMODEL, 1);
    ln_kernel<<<MT, 256, 0, stream>>>(hbuf, nullptr,
                                      ln2_g + l * DMODEL, ln2_b + l * DMODEL, ain, 1);
    gemm_bf16<<<dim3(99, 24), 256, 0, stream>>>(ain, DMODEL,
        wfc1 + (size_t)l * DMODEL * FFDIM, FFDIM,
        fc1_b + (size_t)l * FFDIM, nullptr, nullptr, mlp,
        MT, FFDIM, DMODEL, 2);
    gemm_bf16<<<dim3(99, 6), 256, 0, stream>>>(mlp, FFDIM,
        wfc2 + (size_t)l * FFDIM * DMODEL, DMODEL,
        fc2_b + (size_t)l * DMODEL, hbuf, hbuf, nullptr,
        MT, DMODEL, FFDIM, 1);
  }

  head_kernel<<<BATCH, 256, 0, stream>>>(hbuf, norm_g, norm_b, head_w, head_b, out);
}
